// FlexibleAttention_11854109737109
// MI455X (gfx1250) — matrix operand order repaired and checked by reading, not compiled, not run on hardware
//
#include <hip/hip_runtime.h>

typedef __attribute__((ext_vector_type(16))) _Float16 v16h;
typedef __attribute__((ext_vector_type(8)))  _Float16 v8h;
typedef __attribute__((ext_vector_type(8)))  float    v8f;

#define SEQ    2048
#define DMODEL 1024
#define DHEAD  64
#define NHEADS 16
#define NBH    32          // B * HEADS
#define INNER3 3072

// ---------------------------------------------------------------------------
// CDNA5 async global->LDS copy (ASYNCcnt-tracked), 16 bytes per lane.
// ---------------------------------------------------------------------------
__device__ __forceinline__ void async_copy16(const _Float16* gsrc, _Float16* ldst) {
  unsigned lds_addr = (unsigned)(size_t)ldst;
  unsigned long long gaddr = (unsigned long long)(size_t)gsrc;
  asm volatile("global_load_async_to_lds_b128 %0, %1, off"
               :: "v"(lds_addr), "v"(gaddr) : "memory");
}
__device__ __forceinline__ void wait_async0() {
  asm volatile("s_wait_asynccnt 0" ::: "memory");
}

// ---------------------------------------------------------------------------
// Fragment loaders (CDNA5 wave32 WMMA 16x16x32 f16 layouts, ISA 7.12.2)
// ---------------------------------------------------------------------------
__device__ __forceinline__ v16h frag_ld2(const _Float16* p0, const _Float16* p1) {
  v8h lo = *(const v8h*)p0;
  v8h hi = *(const v8h*)p1;
  v16h r;
#pragma unroll
  for (int i = 0; i < 8; ++i) { r[i] = lo[i]; r[i + 8] = hi[i]; }
  return r;
}

// A fragment: 16(M) x 32(K) tile at (row0,k0) of a row-major [*, ld] f16 matrix.
__device__ __forceinline__ v16h load_a(const _Float16* m, int ld, int row0, int k0, int lane) {
  const _Float16* p = m + (size_t)(row0 + (lane & 15)) * ld + k0 + 8 * (lane >> 4);
  return frag_ld2(p, p + 16);
}

// B fragment: 32(K) x 16(N) where B[k][n] = mT[n][k], mT row-major [*, ld].
__device__ __forceinline__ v16h load_bt(const _Float16* mT, int ld, int n0, int k0, int lane) {
  const _Float16* p = mT + (size_t)(n0 + (lane & 15)) * ld + k0 + 8 * (lane >> 4);
  return frag_ld2(p, p + 16);
}

__device__ __forceinline__ v8f wmma_f16(v16h a, v16h b, v8f c) {
  return __builtin_amdgcn_wmma_f32_16x16x32_f16(false, a, false, b, (short)0, c, false, false);
}

// ---------------------------------------------------------------------------
// fp32 -> f16 conversion kernels
// ---------------------------------------------------------------------------
__global__ void cvt_f32_f16(const float* __restrict__ x, _Float16* __restrict__ xh, int n) {
  int i = blockIdx.x * 256 + threadIdx.x;
  if (i < n) xh[i] = (_Float16)x[i];
}

// w: [rows][cols] fp32 -> wT: [cols][rows] f16
__global__ void cvt_f32_f16_T(const float* __restrict__ w, _Float16* __restrict__ wT,
                              int rows, int cols) {
  int i = blockIdx.x * 256 + threadIdx.x;
  if (i >= rows * cols) return;
  int c = i % cols, r = i / cols;
  wT[(size_t)c * rows + r] = (_Float16)w[i];
}

// ---------------------------------------------------------------------------
// Shared GEMM mainloop: per-wave 32(M) x 64(N) tile, K in steps of 32.
// B tile (64N x 32K = 4 KB) async-staged to LDS, double buffered (pointer
// swap + nounroll keeps WMMA accumulators in place), shared by 8 waves.
// acc[r][t] covers rows m0+16r.., cols n0+16t..
// ---------------------------------------------------------------------------
#define GEMM_MAINLOOP(Aptr, BTptr, ldb)                                          \
  __shared__ __align__(16) _Float16 sB[2][64][32];                               \
  int tid  = threadIdx.x;                                                        \
  int lane = tid & 31;                                                           \
  int wave = tid >> 5;                                                           \
  int m0 = blockIdx.x * 256 + wave * 32;                                         \
  int n0 = blockIdx.y * 64;                                                      \
  int srow = tid >> 2, scol = (tid & 3) * 8;                                     \
  const _Float16* bsrc = (BTptr) + (size_t)(n0 + srow) * (ldb) + scol;           \
  _Float16* cur = &sB[0][0][0];                                                  \
  _Float16* nxt = &sB[1][0][0];                                                  \
  v8f acc[2][4] = {};                                                            \
  async_copy16(bsrc, cur + srow * 32 + scol);                                    \
  _Pragma("clang loop unroll(disable)")                                          \
  for (int it = 0; it < DMODEL / 32; ++it) {                                     \
    int kk = it * 32;                                                            \
    wait_async0();                                                               \
    __syncthreads();                                                             \
    if (it + 1 < DMODEL / 32)                                                    \
      async_copy16(bsrc + kk + 32, nxt + srow * 32 + scol);                      \
    v16h a0 = load_a((Aptr), DMODEL, m0,      kk, lane);                         \
    v16h a1 = load_a((Aptr), DMODEL, m0 + 16, kk, lane);                         \
    v16h b0 = load_bt(cur, 32, 0,  0, lane);                                     \
    v16h b1 = load_bt(cur, 32, 16, 0, lane);                                     \
    v16h b2 = load_bt(cur, 32, 32, 0, lane);                                     \
    v16h b3 = load_bt(cur, 32, 48, 0, lane);                                     \
    acc[0][0] = wmma_f16(a0, b0, acc[0][0]);                                     \
    acc[1][0] = wmma_f16(a1, b0, acc[1][0]);                                     \
    acc[0][1] = wmma_f16(a0, b1, acc[0][1]);                                     \
    acc[1][1] = wmma_f16(a1, b1, acc[1][1]);                                     \
    acc[0][2] = wmma_f16(a0, b2, acc[0][2]);                                     \
    acc[1][2] = wmma_f16(a1, b2, acc[1][2]);                                     \
    acc[0][3] = wmma_f16(a0, b3, acc[0][3]);                                     \
    acc[1][3] = wmma_f16(a1, b3, acc[1][3]);                                     \
    _Float16* tmp = cur; cur = nxt; nxt = tmp;                                   \
  }

// ---------------------------------------------------------------------------
// QKV GEMM: [4096,1024] x [1024,3072] -> scatter into q/k (head-major), v^T
// ---------------------------------------------------------------------------
__global__ void __launch_bounds__(256) gemm_qkv(const _Float16* __restrict__ xh,
                                                const _Float16* __restrict__ wT,
                                                _Float16* __restrict__ qb,
                                                _Float16* __restrict__ kb,
                                                _Float16* __restrict__ vT) {
  GEMM_MAINLOOP(xh, wT, DMODEL)
#pragma unroll
  for (int t = 0; t < 4; ++t) {
    int col   = n0 + 16 * t + (lane & 15);
    int which = col >> 10;                 // 0=q 1=k 2=v
    int inner = col & 1023;
    int h = inner >> 6;
    int d = inner & 63;
#pragma unroll
    for (int r = 0; r < 2; ++r)
#pragma unroll
      for (int vv = 0; vv < 8; ++vv) {
        int row = m0 + 16 * r + vv + 8 * (lane >> 4);  // token index in [0,4096)
        int b   = row >> 11;
        int n   = row & 2047;
        int bh  = (b << 4) + h;
        _Float16 val = (_Float16)acc[r][t][vv];
        if (which == 0)      qb[((size_t)bh * SEQ + n) * DHEAD + d] = val;
        else if (which == 1) kb[((size_t)bh * SEQ + n) * DHEAD + d] = val;
        else                 vT[((size_t)bh * DHEAD + d) * SEQ + n] = val;
      }
  }
}

// ---------------------------------------------------------------------------
// Output GEMM: [4096,1024] x [1024,1024] -> fp32 d_out
// ---------------------------------------------------------------------------
__global__ void __launch_bounds__(256) gemm_out(const _Float16* __restrict__ ah,
                                                const _Float16* __restrict__ wT,
                                                float* __restrict__ out) {
  GEMM_MAINLOOP(ah, wT, DMODEL)
#pragma unroll
  for (int t = 0; t < 4; ++t) {
    int col = n0 + 16 * t + (lane & 15);
#pragma unroll
    for (int r = 0; r < 2; ++r)
#pragma unroll
      for (int vv = 0; vv < 8; ++vv) {
        int row = m0 + 16 * r + vv + 8 * (lane >> 4);
        out[(size_t)row * DMODEL + col] = acc[r][t][vv];
      }
  }
}

// ---------------------------------------------------------------------------
// Per-token 2x2 rotator application on q and k (in place)
// ---------------------------------------------------------------------------
__global__ void rotate_qk(_Float16* __restrict__ qb, _Float16* __restrict__ kb,
                          const float* __restrict__ pos) {
  int t = blockIdx.x * 256 + threadIdx.x;   // total = 2*32*2048*32 = 2^22
  int r  = t & 31;   t >>= 5;
  int n  = t & 2047; t >>= 11;
  int bh = t & 31;   t >>= 5;
  _Float16* buf = t ? kb : qb;
  size_t base = ((size_t)bh * SEQ + n) * DHEAD + 2 * r;
  float a0 = (float)buf[base];
  float a1 = (float)buf[base + 1];
  const float* T = pos + ((size_t)n * 32 + r) * 4;
  buf[base]     = (_Float16)(T[0] * a0 + T[1] * a1);
  buf[base + 1] = (_Float16)(T[2] * a0 + T[3] * a1);
}

// ---------------------------------------------------------------------------
// Flash attention: each wave owns 16 query rows, streams 32 keys per iter.
// K (32x64) and V^T (64x32) tiles async-staged to LDS (double buffered,
// pointer-swapped), shared by the 8 waves of the block.
// ---------------------------------------------------------------------------
__global__ void __launch_bounds__(256) attn_fwd(const _Float16* __restrict__ qb,
                                                const _Float16* __restrict__ kb,
                                                const _Float16* __restrict__ vT,
                                                _Float16* __restrict__ ob) {
  __shared__ __align__(16) _Float16 sk[2][32][64];   // K rows j..j+31
  __shared__ __align__(16) _Float16 sv[2][64][32];   // v^T rows d, cols j..j+31
  __shared__ __align__(16) _Float16 sp[8][16][32];   // per-wave P scratch
  int tid  = threadIdx.x;
  int lane = tid & 31;
  int wave = tid >> 5;
  int bh = blockIdx.x;                       // 0..31
  int m0 = blockIdx.y * 128 + wave * 16;     // query row block
  const _Float16* q = qb + (size_t)bh * SEQ * DHEAD;
  const _Float16* k = kb + (size_t)bh * SEQ * DHEAD;
  const _Float16* v = vT + (size_t)bh * DHEAD * SEQ;
  const float scale = 0.125f;                // DIM_HEAD^-0.5

  // staging geometry: one b128 per thread per tile
  int kr = tid >> 3, kc = (tid & 7) * 8;     // K: 32 rows x 8 chunks
  int vr = tid >> 2, vc = (tid & 3) * 8;     // V: 64 rows x 4 chunks

  v16h qa0 = load_a(q, DHEAD, m0, 0,  lane);
  v16h qa1 = load_a(q, DHEAD, m0, 32, lane);
  v8f o[4] = {};
  float mrow[8], lrow[8];
#pragma unroll
  for (int i = 0; i < 8; ++i) { mrow[i] = -1e30f; lrow[i] = 0.f; }

  _Float16* kcur = &sk[0][0][0]; _Float16* knxt = &sk[1][0][0];
  _Float16* vcur = &sv[0][0][0]; _Float16* vnxt = &sv[1][0][0];
  async_copy16(k + (size_t)kr * DHEAD + kc, kcur + kr * DHEAD + kc);
  async_copy16(v + (size_t)vr * SEQ + vc,   vcur + vr * 32 + vc);
#pragma clang loop unroll(disable)
  for (int j = 0; j < SEQ; j += 32) {
    wait_async0();
    __syncthreads();
    if (j + 32 < SEQ) {
      async_copy16(k + (size_t)(j + 32 + kr) * DHEAD + kc, knxt + kr * DHEAD + kc);
      async_copy16(v + (size_t)vr * SEQ + (j + 32) + vc,   vnxt + vr * 32 + vc);
    }

    // S = Q K^T : two 16x16 tiles (cols j..j+15, j+16..j+31), K-dim = 64
    v8f s0 = {}, s1 = {};
    v16h b00 = load_bt(kcur, DHEAD, 0,  0,  lane);
    v16h b01 = load_bt(kcur, DHEAD, 0,  32, lane);
    v16h b10 = load_bt(kcur, DHEAD, 16, 0,  lane);
    v16h b11 = load_bt(kcur, DHEAD, 16, 32, lane);
    s0 = wmma_f16(qa0, b00, s0);
    s0 = wmma_f16(qa1, b01, s0);
    s1 = wmma_f16(qa0, b10, s1);
    s1 = wmma_f16(qa1, b11, s1);

    // online softmax (rows live across the 16-lane half-wave)
    float alpha[8], mnew[8];
#pragma unroll
    for (int vv = 0; vv < 8; ++vv) {
      s0[vv] *= scale; s1[vv] *= scale;
      float cur = fmaxf(s0[vv], s1[vv]);
      cur = fmaxf(cur, __shfl_xor(cur, 1));
      cur = fmaxf(cur, __shfl_xor(cur, 2));
      cur = fmaxf(cur, __shfl_xor(cur, 4));
      cur = fmaxf(cur, __shfl_xor(cur, 8));
      mnew[vv]  = fmaxf(mrow[vv], cur);
      alpha[vv] = __expf(mrow[vv] - mnew[vv]);
      mrow[vv]  = mnew[vv];
    }
#pragma unroll
    for (int vv = 0; vv < 8; ++vv) {
      float p0 = __expf(s0[vv] - mnew[vv]);
      float p1 = __expf(s1[vv] - mnew[vv]);
      float rs = p0 + p1;
      rs += __shfl_xor(rs, 1);
      rs += __shfl_xor(rs, 2);
      rs += __shfl_xor(rs, 4);
      rs += __shfl_xor(rs, 8);
      lrow[vv] = lrow[vv] * alpha[vv] + rs;
      int row = vv + 8 * (lane >> 4);
      int c   = lane & 15;
      sp[wave][row][c]      = (_Float16)p0;   // C-layout -> LDS
      sp[wave][row][16 + c] = (_Float16)p1;
    }
#pragma unroll
    for (int t = 0; t < 4; ++t)
#pragma unroll
      for (int vv = 0; vv < 8; ++vv) o[t][vv] *= alpha[vv];

    // reload P as A-fragment (16x32), O += P @ V
    v16h pa = load_a(&sp[wave][0][0], 32, 0, 0, lane);
#pragma unroll
    for (int t = 0; t < 4; ++t) {
      v16h vb = load_bt(vcur, 32, 16 * t, 0, lane);
      o[t] = wmma_f16(pa, vb, o[t]);
    }
    _Float16* tk = kcur; kcur = knxt; knxt = tk;
    _Float16* tv = vcur; vcur = vnxt; vnxt = tv;
  }

  // write out in [b, n, h*64+d] layout (f16) for the final GEMM
  int b = bh >> 4, h = bh & 15;
#pragma unroll
  for (int t = 0; t < 4; ++t) {
    int d = 16 * t + (lane & 15);
#pragma unroll
    for (int vv = 0; vv < 8; ++vv) {
      int n = m0 + vv + 8 * (lane >> 4);
      float val = o[t][vv] / lrow[vv];
      ob[((size_t)b * SEQ + n) * DMODEL + h * DHEAD + d] = (_Float16)val;
    }
  }
}

// ---------------------------------------------------------------------------
extern "C" void kernel_launch(void* const* d_in, const int* in_sizes, int n_in,
                              void* d_out, int out_size, void* d_ws, size_t ws_size,
                              hipStream_t stream) {
  const float* x     = (const float*)d_in[0];   // [2,2048,1024]
  const float* pos   = (const float*)d_in[1];   // [2048,32,2,2]
  const float* w_qkv = (const float*)d_in[2];   // [1024,3072]
  const float* w_out = (const float*)d_in[3];   // [1024,1024]
  float* out = (float*)d_out;                   // [2,2048,1024]

  char* ws = (char*)d_ws;
  _Float16* xh    = (_Float16*)(ws);                          // 8 MB
  _Float16* wqkvT = (_Float16*)(ws + ((size_t)8  << 20));     // 6 MB
  _Float16* woutT = (_Float16*)(ws + ((size_t)14 << 20));     // 2 MB
  _Float16* qb    = (_Float16*)(ws + ((size_t)16 << 20));     // 8 MB
  _Float16* kb    = (_Float16*)(ws + ((size_t)24 << 20));     // 8 MB
  _Float16* vT    = (_Float16*)(ws + ((size_t)32 << 20));     // 8 MB
  _Float16* ab    = (_Float16*)(ws + ((size_t)40 << 20));     // 8 MB

  cvt_f32_f16  <<<(4096 * 1024) / 256, 256, 0, stream>>>(x, xh, 4096 * 1024);
  cvt_f32_f16_T<<<(1024 * 3072) / 256, 256, 0, stream>>>(w_qkv, wqkvT, 1024, 3072);
  cvt_f32_f16_T<<<(1024 * 1024) / 256, 256, 0, stream>>>(w_out, woutT, 1024, 1024);

  gemm_qkv<<<dim3(4096 / 256, INNER3 / 64), 256, 0, stream>>>(xh, wqkvT, qb, kb, vT);
  rotate_qk<<<(1 << 22) / 256, 256, 0, stream>>>(qb, kb, pos);
  attn_fwd<<<dim3(NBH, SEQ / 128), 256, 0, stream>>>(qb, kb, vT, ab);
  gemm_out<<<dim3(4096 / 256, DMODEL / 64), 256, 0, stream>>>(ab, woutT, out);
}
